// Model_39676907883957
// MI455X (gfx1250) — matrix operand order repaired and checked by reading, not compiled, not run on hardware
//
#include <hip/hip_runtime.h>

// ---------------------------------------------------------------------------
// Attention-like module for MI455X (gfx1250, wave32, WMMA).
//   q = X Wq^T + bq ; k = ... ; v = ...
//   qk[b,i,j] = sum_t q[b,t,i] * k[b,j,t]   (S == D == 2048)
//   attn = softmax(qk, axis=-1) ; out[b,d,i] = sum_j attn[i,j] v[b,j,d]
// All GEMMs run on v_wmma_f32_16x16x32_bf16 (f32 accumulate), softmax in f32.
// Wave tile = 32(M) x 64(N): 2 A frags x 4 B frags -> 8 WMMAs per K-step.
// Block = 8 waves as 2(M) x 4(N) -> 64 x 256 block tile.
// ---------------------------------------------------------------------------

typedef __attribute__((ext_vector_type(16))) __bf16 v16bf;
typedef __attribute__((ext_vector_type(8)))  __bf16 v8bf;
typedef __attribute__((ext_vector_type(8)))  float  v8f;

static constexpr int Bn = 8;     // batch
static constexpr int S  = 2048;  // sequence
static constexpr int D  = 2048;  // features (== S)

// ---- WMMA fragment loaders (contiguous k order, shared by A and B) -------------------
// A 16x32 bf16: lane l -> row (l&15); h=l>>4; elems i = K(16h + i)
__device__ __forceinline__ v16bf load_a_bf16(const __bf16* __restrict__ row,
                                             int kk, int h) {
  v8bf c0 = *(const v8bf*)(row + kk + 16 * h);
  v8bf c1 = *(const v8bf*)(row + kk + 16 * h + 8);
  v16bf a;
#pragma unroll
  for (int i = 0; i < 8; ++i) { a[i] = c0[i]; a[i + 8] = c1[i]; }
  return a;
}
// B 32x16 bf16: lane l -> col (l&15); elems i = K(16h + i): one contiguous run
__device__ __forceinline__ v16bf load_b_bf16(const __bf16* __restrict__ col,
                                             int kk, int h) {
  return *(const v16bf*)(col + kk + 16 * h);
}
// Same fragments but sourced from fp32 with in-register cvt to bf16
__device__ __forceinline__ v16bf cvt_a_f32(const float* __restrict__ row,
                                           int kk, int h) {
  v8f c0 = *(const v8f*)(row + kk + 16 * h);
  v8f c1 = *(const v8f*)(row + kk + 16 * h + 8);
  v16bf a;
#pragma unroll
  for (int i = 0; i < 8; ++i) { a[i] = (__bf16)c0[i]; a[i + 8] = (__bf16)c1[i]; }
  return a;
}
__device__ __forceinline__ v16bf cvt_b_f32(const float* __restrict__ wrow,
                                           int kk, int h) {
  v8f c0 = *(const v8f*)(wrow + kk + 16 * h);
  v8f c1 = *(const v8f*)(wrow + kk + 16 * h + 8);
  v16bf b;
#pragma unroll
  for (int i = 0; i < 8; ++i) { b[i] = (__bf16)c0[i]; b[i + 8] = (__bf16)c1[i]; }
  return b;
}

// ---- Projection: Y = X @ W^T + bias, fp32 in -> bf16 out -------------------
// TRANSPOSED==0: Y[b*S+s][e] ([B][S][D]); ==1: Y[b][e][s] ([B][D][S]).
template <int TRANSPOSED>
__global__ __launch_bounds__(256)
void proj_gemm_kernel(const float* __restrict__ X, const float* __restrict__ W,
                      const float* __restrict__ bias, __bf16* __restrict__ Y) {
  const int lane = threadIdx.x & 31;
  const int wave = threadIdx.x >> 5;
  const int r = lane & 15, h = lane >> 4;
  const int m0 = blockIdx.x * 64 + (wave >> 2) * 32;
  const int n0 = blockIdx.y * 256 + (wave & 3) * 64;
  const float* arow[2];
#pragma unroll
  for (int u = 0; u < 2; ++u)
    arow[u] = X + (size_t)(m0 + 16 * u + r) * D;    // A rows for this lane
  const float* wrow[4];
#pragma unroll
  for (int t = 0; t < 4; ++t)
    wrow[t] = W + (size_t)(n0 + 16 * t + r) * D;    // W row == B column
  v8f acc[2][4] = {};
  for (int kk = 0; kk < D; kk += 32) {
    __builtin_prefetch(arow[0] + kk + 128, 0, 1);
    v16bf a[2];
#pragma unroll
    for (int u = 0; u < 2; ++u) a[u] = cvt_a_f32(arow[u], kk, h);
#pragma unroll
    for (int t = 0; t < 4; ++t) {
      const v16bf b = cvt_b_f32(wrow[t], kk, h);
#pragma unroll
      for (int u = 0; u < 2; ++u)
        acc[u][t] = __builtin_amdgcn_wmma_f32_16x16x32_bf16(
            false, a[u], false, b, (short)0, acc[u][t], false, false);
    }
  }
#pragma unroll
  for (int t = 0; t < 4; ++t) {
    const int n = n0 + 16 * t + r;     // output feature (N depends on lane)
    const float bv = bias[n];
#pragma unroll
    for (int u = 0; u < 2; ++u)
#pragma unroll
      for (int rr = 0; rr < 8; ++rr) {
        const int m = m0 + 16 * u + rr + 8 * h;   // global row (b*S + s)
        size_t idx;
        if (TRANSPOSED) {
          const int bb = m >> 11, ss = m & (S - 1);   // S = 2048
          idx = ((size_t)bb * D + n) * S + ss;
        } else {
          idx = (size_t)m * D + n;
        }
        Y[idx] = (__bf16)(acc[u][t][rr] + bv);
      }
  }
}

// ---- Logits: L[i][j] = sum_t qT[i][t] * k[j][t] (one batch) ----------------
__global__ __launch_bounds__(256)
void logits_kernel(const __bf16* __restrict__ qT,   // [D][S]
                   const __bf16* __restrict__ Kp,   // [S][D], row j = B col j
                   float* __restrict__ L) {         // [D][S]
  const int lane = threadIdx.x & 31;
  const int wave = threadIdx.x >> 5;
  const int r = lane & 15, h = lane >> 4;
  const int m0 = blockIdx.x * 64 + (wave >> 2) * 32;
  const int n0 = blockIdx.y * 256 + (wave & 3) * 64;
  const __bf16* arow[2];
#pragma unroll
  for (int u = 0; u < 2; ++u) arow[u] = qT + (size_t)(m0 + 16 * u + r) * S;
  const __bf16* bcol[4];
#pragma unroll
  for (int t = 0; t < 4; ++t) bcol[t] = Kp + (size_t)(n0 + 16 * t + r) * D;
  v8f acc[2][4] = {};
  for (int kk = 0; kk < S; kk += 32) {
    __builtin_prefetch(arow[0] + kk + 256, 0, 1);
    v16bf a[2];
#pragma unroll
    for (int u = 0; u < 2; ++u) a[u] = load_a_bf16(arow[u], kk, h);
#pragma unroll
    for (int t = 0; t < 4; ++t) {
      const v16bf b = load_b_bf16(bcol[t], kk, h);
#pragma unroll
      for (int u = 0; u < 2; ++u)
        acc[u][t] = __builtin_amdgcn_wmma_f32_16x16x32_bf16(
            false, a[u], false, b, (short)0, acc[u][t], false, false);
    }
  }
#pragma unroll
  for (int t = 0; t < 4; ++t)
#pragma unroll
    for (int u = 0; u < 2; ++u)
#pragma unroll
      for (int rr = 0; rr < 8; ++rr)
        L[(size_t)(m0 + 16 * u + rr + 8 * h) * S + (n0 + 16 * t + r)] =
            acc[u][t][rr];  // scale_factor = 1
}

// ---- Row softmax: f32 logits -> bf16 attn, one block per row ---------------
__global__ __launch_bounds__(256)
void softmax_kernel(const float* __restrict__ L, __bf16* __restrict__ A) {
  const int row = blockIdx.x, tid = threadIdx.x;
  const float* x = L + (size_t)row * S;
  __bf16* y = A + (size_t)row * S;
  __shared__ float red[256];
  float v[8];
  float lmax = -3.4e38f;
#pragma unroll
  for (int k = 0; k < 8; ++k) { v[k] = x[tid + k * 256]; lmax = fmaxf(lmax, v[k]); }
  red[tid] = lmax; __syncthreads();
  for (int s2 = 128; s2 > 0; s2 >>= 1) {
    if (tid < s2) red[tid] = fmaxf(red[tid], red[tid + s2]);
    __syncthreads();
  }
  const float m = red[0]; __syncthreads();
  float lsum = 0.f;
#pragma unroll
  for (int k = 0; k < 8; ++k) { v[k] = __expf(v[k] - m); lsum += v[k]; }
  red[tid] = lsum; __syncthreads();
  for (int s2 = 128; s2 > 0; s2 >>= 1) {
    if (tid < s2) red[tid] += red[tid + s2];
    __syncthreads();
  }
  const float inv = 1.0f / red[0];
#pragma unroll
  for (int k = 0; k < 8; ++k) y[tid + k * 256] = (__bf16)(v[k] * inv);
}

// ---- Output: out[d][i] = sum_j attn[i][j] * vT[d][j] (one batch) -----------
// (final transpose of the reference folded into the store)
__global__ __launch_bounds__(256)
void out_gemm_kernel(const __bf16* __restrict__ attn,  // [D][S]
                     const __bf16* __restrict__ vT,    // [D][S], row d = B col d
                     float* __restrict__ out) {        // [D][D], [d][i]
  const int lane = threadIdx.x & 31;
  const int wave = threadIdx.x >> 5;
  const int r = lane & 15, h = lane >> 4;
  const int m0 = blockIdx.x * 64 + (wave >> 2) * 32;   // i
  const int n0 = blockIdx.y * 256 + (wave & 3) * 64;   // d
  const __bf16* arow[2];
#pragma unroll
  for (int u = 0; u < 2; ++u) arow[u] = attn + (size_t)(m0 + 16 * u + r) * S;
  const __bf16* bcol[4];
#pragma unroll
  for (int t = 0; t < 4; ++t) bcol[t] = vT + (size_t)(n0 + 16 * t + r) * S;
  v8f acc[2][4] = {};
  for (int kk = 0; kk < S; kk += 32) {
    __builtin_prefetch(arow[0] + kk + 256, 0, 1);
    v16bf a[2];
#pragma unroll
    for (int u = 0; u < 2; ++u) a[u] = load_a_bf16(arow[u], kk, h);
#pragma unroll
    for (int t = 0; t < 4; ++t) {
      const v16bf b = load_b_bf16(bcol[t], kk, h);
#pragma unroll
      for (int u = 0; u < 2; ++u)
        acc[u][t] = __builtin_amdgcn_wmma_f32_16x16x32_bf16(
            false, a[u], false, b, (short)0, acc[u][t], false, false);
    }
  }
#pragma unroll
  for (int t = 0; t < 4; ++t)
#pragma unroll
    for (int u = 0; u < 2; ++u)
#pragma unroll
      for (int rr = 0; rr < 8; ++rr)
        out[(size_t)(n0 + 16 * t + r) * D + (m0 + 16 * u + rr + 8 * h)] =
            acc[u][t][rr];
}

// ---------------------------------------------------------------------------
extern "C" void kernel_launch(void* const* d_in, const int* in_sizes, int n_in,
                              void* d_out, int out_size, void* d_ws,
                              size_t ws_size, hipStream_t stream) {
  (void)in_sizes; (void)n_in; (void)out_size; (void)ws_size;
  const float* query = (const float*)d_in[0];
  const float* key_  = (const float*)d_in[1];
  const float* value = (const float*)d_in[2];
  const float* Wq = (const float*)d_in[3];
  const float* bq = (const float*)d_in[4];
  const float* Wk = (const float*)d_in[5];
  const float* bk = (const float*)d_in[6];
  const float* Wv = (const float*)d_in[7];
  const float* bv = (const float*)d_in[8];
  float* out = (float*)d_out;

  // Workspace layout (~227 MB):
  //   qT  bf16 [B][D][S]   kprj bf16 [B][S][D]   vT bf16 [B][D][S]
  //   Lg  f32  [D][S] (reused per batch)         At bf16 [D][S]
  char* ws = (char*)d_ws;
  const size_t mat = (size_t)Bn * S * D;
  __bf16* qT   = (__bf16*)ws;
  __bf16* kprj = (__bf16*)(ws + mat * 2);
  __bf16* vT   = (__bf16*)(ws + mat * 4);
  float*  Lg   = (float*) (ws + mat * 6);
  __bf16* At   = (__bf16*)(ws + mat * 6 + (size_t)D * S * 4);

  dim3 blk(256);
  dim3 gp((Bn * S) / 64, D / 256);
  proj_gemm_kernel<1><<<gp, blk, 0, stream>>>(query, Wq, bq, qT);
  proj_gemm_kernel<0><<<gp, blk, 0, stream>>>(key_,  Wk, bk, kprj);
  proj_gemm_kernel<1><<<gp, blk, 0, stream>>>(value, Wv, bv, vT);

  dim3 ga(D / 64, S / 256);
  for (int b = 0; b < Bn; ++b) {
    const __bf16* qTb = qT + (size_t)b * D * S;
    const __bf16* kb  = kprj + (size_t)b * S * D;
    const __bf16* vTb = vT + (size_t)b * D * S;
    logits_kernel<<<ga, blk, 0, stream>>>(qTb, kb, Lg);
    softmax_kernel<<<dim3(D), blk, 0, stream>>>(Lg, At);
    out_gemm_kernel<<<ga, blk, 0, stream>>>(At, vTb, out + (size_t)b * D * D);
  }
}
